// NEQUIPESCNLayerFlax_59365037965772
// MI455X (gfx1250) — matrix operand order repaired and checked by reading, not compiled, not run on hardware
//
#include <hip/hip_runtime.h>
#include <hip/hip_bf16.h>

// ---------------------------------------------------------------------------
// NequIP/ESCN-style equivariant GNN layer for gfx1250 (MI455X).
//   - equivariant GEMMs (up-proj, Wd0/Wd1): V_WMMA_F32_16X16X4_F32 (fp32)
//   - dominant radial MLP (16 of 19 GFLOP): V_WMMA_F32_16X16X32_BF16
//   - phase4/5 fused: gate WMMA accumulators consumed directly by the
//     message/scatter code (no wmix LDS round-trip)
//   - u1/a1 in planar (N,3,C) workspace layout: coalesced gathers/atomics
// ---------------------------------------------------------------------------

#define C 64
#define NB 8
#define NP 5
#define TE 64            // edges per workgroup in edge kernel
#define INV_SQRT_AVG 0.25f
#define PI_F 3.14159265358979323846f

typedef __attribute__((ext_vector_type(2)))  float  v2f;
typedef __attribute__((ext_vector_type(8)))  float  v8f;
typedef __attribute__((ext_vector_type(16))) __bf16 v16bf;

union BfFrag { v16bf v; unsigned int u[8]; uint4 q[2]; };

__device__ __forceinline__ v8f wmma_f32_k4(v2f a, v2f b, v8f c) {
  return __builtin_amdgcn_wmma_f32_16x16x4_f32(false, a, false, b, (short)0, c,
                                               false, false);
}
__device__ __forceinline__ v8f wmma_bf16_k32(v16bf a, v16bf b, v8f c) {
  return __builtin_amdgcn_wmma_f32_16x16x32_bf16(false, a, false, b, (short)0, c,
                                                 false, false);
}

__device__ __forceinline__ float silu(float x) {
  return x * (1.0f / (1.0f + __expf(-x)));
}

__device__ __forceinline__ unsigned short f32_to_bf16(float x) {
  unsigned int u = __float_as_uint(x);
  unsigned int r = (u + 0x7FFFu + ((u >> 16) & 1u)) >> 16;  // RNE
  return (unsigned short)r;
}

// A fragment, 16-bit 16x32 (ISA 7.12.2): lane M = lane&15; VGPR v holds
// K = (v<4?0:16) + 8*half + 2*(v&3) + {0,1}. Row-major bf16 src, ld elements.
__device__ __forceinline__ v16bf load_a_bf16(const unsigned short* s, int ld,
                                             int row, int k0, int half) {
  const unsigned short* p = s + row * ld + k0 + 8 * half;
  BfFrag f;
  f.q[0] = *(const uint4*)(p);       // K pairs 0,2,4,6   (+8*half)
  f.q[1] = *(const uint4*)(p + 16);  // K pairs 16,18,20,22
  return f.v;
}

// B fragment, 16-bit 32x16: lane N = lane&15; element e holds K = (e<8?0:16) + 8*half + (e&7).
// Source is K-contiguous transposed weight Wt[n][k], ldk elements.
__device__ __forceinline__ v16bf load_b_bf16(const unsigned short* Wt, int ldk,
                                             int n, int k0, int half) {
  const unsigned short* p = Wt + (size_t)n * ldk + k0 + 8 * half;
  BfFrag f;
  f.q[0] = *(const uint4*)(p);
  f.q[1] = *(const uint4*)(p + 16);
  return f.v;
}

// ---------------------------------------------------------------------------
// Kernel 0: transpose + convert radial-MLP weights to bf16 (K-contiguous).
// ---------------------------------------------------------------------------
__global__ void convert_weights_kernel(const float* __restrict__ Wm1,
                                       const float* __restrict__ Wm2,
                                       const float* __restrict__ Wm3,
                                       unsigned short* __restrict__ Wm1t,
                                       unsigned short* __restrict__ Wm2t,
                                       unsigned short* __restrict__ Wm3t) {
  const int tid = blockIdx.x * blockDim.x + threadIdx.x;
  const int stride = gridDim.x * blockDim.x;
  for (int p = tid; p < C * 32; p += stride) {
    const int n = p >> 5, k = p & 31;
    Wm1t[p] = (k < NB) ? f32_to_bf16(Wm1[k * C + n]) : (unsigned short)0;
  }
  for (int p = tid; p < C * C; p += stride) {
    const int n = p >> 6, k = p & 63;
    Wm2t[p] = f32_to_bf16(Wm2[k * C + n]);
  }
  for (int p = tid; p < NP * C * C; p += stride) {
    const int n = p >> 6, k = p & 63;
    Wm3t[p] = f32_to_bf16(Wm3[k * (NP * C) + n]);
  }
}

// ---------------------------------------------------------------------------
// Kernel 1: u0 = x0 @ W0_up ; u1[i] = x1[:,:,i] @ W1_up (planar out, fp32 WMMA)
// ---------------------------------------------------------------------------
__global__ void up_proj_kernel(const float* __restrict__ nf,
                               const float* __restrict__ W0,
                               const float* __restrict__ W1,
                               float* __restrict__ u0, float* __restrict__ u1,
                               int N) {
  extern __shared__ float sm_up[];
  float* s_nf = sm_up;               // 32*256
  float* s_w0 = s_nf + 32 * 4 * C;   // 64*64
  float* s_w1 = s_w0 + C * C;        // 64*64

  const int tid = threadIdx.x;
  const int n0 = blockIdx.x * 32;

  for (int p = tid; p < 32 * C; p += 256) {       // 2048 float4, coalesced
    const int r = p >> 6, q4 = p & 63;
    int row = n0 + r;
    if (row >= N) row = N - 1;
    ((float4*)s_nf)[p] = ((const float4*)(nf + (size_t)row * (4 * C)))[q4];
  }
  for (int p = tid; p < C * 16; p += 256) {       // 1024 float4 each matrix
    ((float4*)s_w0)[p] = ((const float4*)W0)[p];
    ((float4*)s_w1)[p] = ((const float4*)W1)[p];
  }
  __syncthreads();

  const int wave = tid >> 5, lane = tid & 31, hl = lane >> 4, l16 = lane & 15;

  for (int t = wave; t < 32; t += 8) {
    const int rloc = (t & 1) * 16;
    const int mat = (t >> 1) & 3;                 // 0: x0*W0 ; 1..3: x1_i*W1
    const int c0 = (t >> 3) * 16;
    const float* sW   = (mat == 0) ? s_w0 : s_w1;
    const int   stepk = (mat == 0) ? 1 : 3;       // branchless A addressing
    const int   offk  = (mat == 0) ? 0 : (C + (mat - 1));
    const float* pa0 = s_nf + (rloc + l16) * (4 * C) + offk;

    v8f acc = {};
    for (int k = 0; k < C; k += 4) {
      const int ka = k + 2 * hl;
      v2f a, b;
      a.x = pa0[ka * stepk];
      a.y = pa0[(ka + 1) * stepk];
      const float* q = sW + ka * C + c0 + l16;
      b.x = q[0]; b.y = q[C];
      acc = wmma_f32_k4(a, b, acc);
    }
    const int d = c0 + l16;
    if (mat == 0) {
#pragma unroll
      for (int j = 0; j < 8; ++j) {
        const int n = n0 + rloc + j + 8 * hl;
        if (n < N) u0[(size_t)n * C + d] = acc[j];
      }
    } else {
#pragma unroll
      for (int j = 0; j < 8; ++j) {
        const int n = n0 + rloc + j + 8 * hl;
        if (n < N) u1[((size_t)n * 3 + (mat - 1)) * C + d] = acc[j];  // planar
      }
    }
  }
}

// ---------------------------------------------------------------------------
// Kernel 2: fused edge pipeline (block = 256 threads, TE = 64 edges, ~22 KB LDS)
//   phase1: r, Y1, radial basis -> bf16 (K padded to 32)
//   phase2: H1 = silu(basis @ Wm1)  1 bf16-WMMA / tile
//   phase3: H2 = silu(H1 @ Wm2)     2 bf16-WMMA / tile
//   phase4+5 FUSED: per (row-tile, 16-chan block), one wave computes all 5
//     gate tiles (10 bf16-WMMA, A reused), then builds messages from its own
//     accumulators and scatters with global_atomic_add_f32.
// ---------------------------------------------------------------------------
__global__ void edge_kernel(const float* __restrict__ vectors,
                            const int* __restrict__ senders,
                            const int* __restrict__ receivers,
                            const unsigned short* __restrict__ Wm1t,
                            const unsigned short* __restrict__ Wm2t,
                            const unsigned short* __restrict__ Wm3t,
                            const float* __restrict__ u0,
                            const float* __restrict__ u1,
                            float* __restrict__ a0, float* __restrict__ a1,
                            int E) {
  extern __shared__ char smem[];
  float*          s_y     = (float*)smem;                   // TE*3
  float*          s_r     = s_y + TE * 3;                   // TE
  int*            s_send  = (int*)(s_r + TE);               // TE
  int*            s_recv  = s_send + TE;                    // TE
  unsigned short* s_basis = (unsigned short*)(s_recv + TE); // TE*32 bf16
  unsigned short* s_h1    = s_basis + TE * 32;              // TE*64 bf16
  unsigned short* s_h2    = s_h1 + TE * C;                  // TE*64 bf16

  const int tid = threadIdx.x;
  const int e0 = blockIdx.x * TE;

  if (tid == 0) __builtin_prefetch(Wm3t, 0, 1);

  // ---- phase 1: geometry + radial basis ----
  if (tid < TE) {
    const int e = e0 + tid;
    if (e < E) {
      const float vx = vectors[(size_t)e * 3 + 0];
      const float vy = vectors[(size_t)e * 3 + 1];
      const float vz = vectors[(size_t)e * 3 + 2];
      const float r = sqrtf(vx * vx + vy * vy + vz * vz);
      const float safe = (r == 0.0f) ? 1.0f : r;
      const float rinv = 1.0f / safe;
      s_r[tid] = r;
      s_y[tid * 3 + 0] = vx * rinv;
      s_y[tid * 3 + 1] = vy * rinv;
      s_y[tid * 3 + 2] = vz * rinv;
      const float scale = sqrtf(2.0f / 5.0f) * rinv;
      const float w = (PI_F / 5.0f) * r;
#pragma unroll
      for (int k = 0; k < 32; ++k)
        s_basis[tid * 32 + k] =
            (k < NB) ? f32_to_bf16(scale * __sinf((float)(k + 1) * w))
                     : (unsigned short)0;
      s_send[tid] = senders[e];
      s_recv[tid] = receivers[e];
    } else {
      s_r[tid] = 1.0f;
      s_y[tid * 3 + 0] = 0.f; s_y[tid * 3 + 1] = 0.f; s_y[tid * 3 + 2] = 0.f;
#pragma unroll
      for (int k = 0; k < 32; ++k) s_basis[tid * 32 + k] = 0;
      s_send[tid] = 0;
      s_recv[tid] = -1;
    }
  }
  __syncthreads();

  const int wave = tid >> 5, lane = tid & 31, hl = lane >> 4, l16 = lane & 15;

  // ---- phase 2: H1 = silu(basis @ Wm1) : 16 tiles, K=32 (8 + zero pad) ----
  for (int t = wave; t < 16; t += 8) {
    const int rt = (t >> 2) * 16, ct = (t & 3) * 16;
    v16bf a = load_a_bf16(s_basis, 32, rt + l16, 0, hl);
    v16bf b = load_b_bf16(Wm1t, 32, ct + l16, 0, hl);
    v8f acc = {};
    acc = wmma_bf16_k32(a, b, acc);
#pragma unroll
    for (int j = 0; j < 8; ++j)
      s_h1[(rt + j + 8 * hl) * C + ct + l16] = f32_to_bf16(silu(acc[j]));
  }
  __syncthreads();

  // ---- phase 3: H2 = silu(H1 @ Wm2) : 16 tiles, K=64 ----
  for (int t = wave; t < 16; t += 8) {
    const int rt = (t >> 2) * 16, ct = (t & 3) * 16;
    v8f acc = {};
#pragma unroll
    for (int k0 = 0; k0 < C; k0 += 32) {
      v16bf a = load_a_bf16(s_h1, C, rt + l16, k0, hl);
      v16bf b = load_b_bf16(Wm2t, C, ct + l16, k0, hl);
      acc = wmma_bf16_k32(a, b, acc);
    }
#pragma unroll
    for (int j = 0; j < 8; ++j)
      s_h2[(rt + j + 8 * hl) * C + ct + l16] = f32_to_bf16(silu(acc[j]));
  }
  __syncthreads();

  // ---- phase 4+5 fused: 16 tile-sets (4 row-tiles x 4 channel blocks) ----
  for (int s = wave; s < 16; s += 8) {
    const int rt = (s >> 2) * 16;          // edge-row tile
    const int cb = (s & 3) * 16;           // channel block within C
    // A fragments loaded once, reused for the 5 path tiles
    const v16bf a0f = load_a_bf16(s_h2, C, rt + l16, 0, hl);
    const v16bf a1f = load_a_bf16(s_h2, C, rt + l16, 32, hl);
    v8f acc[NP];
#pragma unroll
    for (int pth = 0; pth < NP; ++pth) {
      const int n = pth * C + cb + l16;    // column in Wm3t (320 rows)
      v16bf b0 = load_b_bf16(Wm3t, C, n, 0, hl);
      v16bf b1 = load_b_bf16(Wm3t, C, n, 32, hl);
      v8f t = {};
      t = wmma_bf16_k32(a0f, b0, t);
      t = wmma_bf16_k32(a1f, b1, t);
      acc[pth] = t;
    }
    const int c = cb + l16;
#pragma unroll
    for (int j = 0; j < 8; ++j) {
      const int le = rt + j + 8 * hl;      // local edge index
      const int rcv = s_recv[le];
      if (rcv < 0) continue;
      const float rz = (s_r[le] == 0.0f) ? 0.0f : 1.0f;
      const float w00  = acc[0][j] * rz;
      const float w01  = acc[1][j] * rz;
      const float w10  = acc[2][j] * rz;
      const float w11d = acc[3][j] * rz;
      const float w11c = acc[4][j] * rz;
      const int snd = s_send[le];
      const float m0  = u0[(size_t)snd * C + c];
      const float m1x = u1[((size_t)snd * 3 + 0) * C + c];
      const float m1y = u1[((size_t)snd * 3 + 1) * C + c];
      const float m1z = u1[((size_t)snd * 3 + 2) * C + c];
      const float yx = s_y[le * 3 + 0], yy = s_y[le * 3 + 1], yz = s_y[le * 3 + 2];
      const float dot = m1x * yx + m1y * yy + m1z * yz;
      const float cx = m1y * yz - m1z * yy;
      const float cy = m1z * yx - m1x * yz;
      const float cz = m1x * yy - m1y * yx;
      const float msg0 = (w00 * m0 + w11d * dot) * INV_SQRT_AVG;
      const float g = w01 * m0;
      const float o0 = (g * yx + w10 * m1x + w11c * cx) * INV_SQRT_AVG;
      const float o1 = (g * yy + w10 * m1y + w11c * cy) * INV_SQRT_AVG;
      const float o2 = (g * yz + w10 * m1z + w11c * cz) * INV_SQRT_AVG;
      atomicAdd(&a0[(size_t)rcv * C + c], msg0);
      atomicAdd(&a1[((size_t)rcv * 3 + 0) * C + c], o0);
      atomicAdd(&a1[((size_t)rcv * 3 + 1) * C + c], o1);
      atomicAdd(&a1[((size_t)rcv * 3 + 2) * C + c], o2);
    }
  }
}

// ---------------------------------------------------------------------------
// Kernel 3: node output (fp32 WMMA for Wd0/Wd1, VALU species matvecs, gating).
// ---------------------------------------------------------------------------
__global__ void node_out_kernel(const float* __restrict__ nf,
                                const int* __restrict__ specie,
                                const float* __restrict__ a0,
                                const float* __restrict__ a1,
                                const float* __restrict__ Wd0,
                                const float* __restrict__ Wd1,
                                const float* __restrict__ Ws0,
                                const float* __restrict__ Ws1,
                                float* __restrict__ out, int N) {
  __shared__ float s_x0[16 * C];
  __shared__ float s_x1[16 * 3 * C];    // [n][c][i] (reference layout)
  __shared__ float s_f0[16 * 2 * C];
  __shared__ float s_d1[16 * 3 * C];    // [n][d][i]
  __shared__ int s_sp[16];

  const int tid = threadIdx.x;
  const int n0 = blockIdx.x * 16;

  for (int p = tid; p < 16 * 4 * C; p += 256) {
    const int n = p >> 8;
    const int c = p & (4 * C - 1);
    const int nn = n0 + n;
    const float v = (nn < N) ? nf[(size_t)nn * (4 * C) + c] : 0.f;
    if (c < C) s_x0[n * C + c] = v;
    else       s_x1[n * (3 * C) + (c - C)] = v;
  }
  if (tid < 16) {
    const int nn = n0 + tid;
    s_sp[tid] = (nn < N) ? specie[nn] : 0;
  }

  const int wave = tid >> 5, lane = tid & 31, hl = lane >> 4, l16 = lane & 15;
  int row = n0 + l16;
  if (row >= N) row = N - 1;

  for (int t = wave; t < 20; t += 8) {
    v8f acc = {};
    if (t < 8) {
      const int c0 = t * 16;
      for (int k = 0; k < C; k += 4) {
        const int ka = k + 2 * hl;
        v2f a, b;
        const float* pa = a0 + (size_t)row * C + ka;
        a.x = pa[0]; a.y = pa[1];
        const float* pb = Wd0 + (size_t)ka * (2 * C) + c0 + l16;
        b.x = pb[0]; b.y = pb[2 * C];
        acc = wmma_f32_k4(a, b, acc);
      }
#pragma unroll
      for (int j = 0; j < 8; ++j)
        s_f0[(j + 8 * hl) * (2 * C) + c0 + l16] = acc[j];
    } else {
      const int tt = t - 8;
      const int i = tt >> 2;
      const int c0 = (tt & 3) * 16;
      for (int k = 0; k < C; k += 4) {
        const int ka = k + 2 * hl;
        v2f a, b;
        const float* pa = a1 + ((size_t)row * 3 + i) * C + ka;  // planar
        a.x = pa[0]; a.y = pa[1];
        const float* pb = Wd1 + (size_t)ka * C + c0 + l16;
        b.x = pb[0]; b.y = pb[C];
        acc = wmma_f32_k4(a, b, acc);
      }
#pragma unroll
      for (int j = 0; j < 8; ++j)
        s_d1[(j + 8 * hl) * (3 * C) + (c0 + l16) * 3 + i] = acc[j];
    }
  }
  __syncthreads();

  for (int p = tid; p < 16 * 2 * C; p += 256) {
    const int n = p >> 7;
    const int d = p & (2 * C - 1);
    const int sp = s_sp[n];
    const float* Wp = Ws0 + (size_t)sp * C * 2 * C + d;
    float sum = 0.f;
#pragma unroll 8
    for (int c = 0; c < C; ++c) sum += s_x0[n * C + c] * Wp[(size_t)c * (2 * C)];
    s_f0[p] = silu(s_f0[p] + sum);
  }
  __syncthreads();

  const int nvalid = (N - n0 < 16) ? (N - n0) : 16;

  for (int p = tid; p < 16 * C; p += 256) {
    const int n = p >> 6;
    const int d = p & (C - 1);
    if (n < nvalid) out[(size_t)(n0 + n) * (4 * C) + d] = s_f0[n * (2 * C) + d];
  }
  for (int p = tid; p < 16 * 3 * C; p += 256) {
    const int n = p / (3 * C);
    const int di = p % (3 * C);
    const int d = di / 3;
    const int i = di % 3;
    if (n >= nvalid) continue;
    const int sp = s_sp[n];
    const float* Wp = Ws1 + (size_t)sp * C * C + d;
    float sum = 0.f;
#pragma unroll 8
    for (int c = 0; c < C; ++c) sum += s_x1[n * (3 * C) + c * 3 + i] * Wp[(size_t)c * C];
    const float f1 = s_d1[p] + sum;
    const float gate = s_f0[n * (2 * C) + C + d];
    out[(size_t)(n0 + n) * (4 * C) + C + di] = f1 * gate;
  }
}

// ---------------------------------------------------------------------------
extern "C" void kernel_launch(void* const* d_in, const int* in_sizes, int n_in,
                              void* d_out, int out_size, void* d_ws, size_t ws_size,
                              hipStream_t stream) {
  const float* vectors   = (const float*)d_in[0];
  const float* nf        = (const float*)d_in[1];
  const int*   specie    = (const int*)d_in[2];
  const int*   senders   = (const int*)d_in[3];
  const int*   receivers = (const int*)d_in[4];
  const float* W0_up     = (const float*)d_in[5];
  const float* W1_up     = (const float*)d_in[6];
  const float* Wm1       = (const float*)d_in[7];
  const float* Wm2       = (const float*)d_in[8];
  const float* Wm3       = (const float*)d_in[9];
  const float* Ws0       = (const float*)d_in[10];
  const float* Ws1       = (const float*)d_in[11];
  const float* Wd0       = (const float*)d_in[12];
  const float* Wd1       = (const float*)d_in[13];

  const int N = in_sizes[1] / (4 * C);
  const int E = in_sizes[0] / 3;

  float* u0 = (float*)d_ws;                  // N x 64
  float* u1 = u0 + (size_t)N * C;            // N x 3 x 64 (planar)
  float* a0 = u1 + (size_t)N * 3 * C;        // N x 64
  float* a1 = a0 + (size_t)N * C;            // N x 3 x 64 (planar)
  unsigned short* Wm1t = (unsigned short*)(a1 + (size_t)N * 3 * C); // 64x32
  unsigned short* Wm2t = Wm1t + C * 32;                             // 64x64
  unsigned short* Wm3t = Wm2t + C * C;                              // 320x64

  hipMemsetAsync(a0, 0, (size_t)N * 4 * C * sizeof(float), stream);

  convert_weights_kernel<<<40, 256, 0, stream>>>(Wm1, Wm2, Wm3, Wm1t, Wm2t, Wm3t);

  const size_t lds_up = (size_t)(32 * 4 * C + 2 * C * C) * sizeof(float); // 64 KB
  up_proj_kernel<<<(N + 31) / 32, 256, lds_up, stream>>>(nf, W0_up, W1_up, u0, u1, N);

  const size_t lds_edge =
      (size_t)(TE * 3 + TE + TE + TE) * sizeof(float) +                // y, r, send, recv
      (size_t)(TE * 32 + TE * C + TE * C) * sizeof(unsigned short);    // basis, h1, h2
  edge_kernel<<<(E + TE - 1) / TE, 256, lds_edge, stream>>>(
      vectors, senders, receivers, Wm1t, Wm2t, Wm3t, u0, u1, a0, a1, E);

  node_out_kernel<<<(N + 15) / 16, 256, 0, stream>>>(
      nf, specie, a0, a1, Wd0, Wd1, Ws0, Ws1, (float*)d_out, N);
}
